// MultiHeadAttention_10728828305608
// MI455X (gfx1250) — matrix operand order repaired and checked by reading, not compiled, not run on hardware
//
#include <hip/hip_runtime.h>
#include <hip/hip_bf16.h>

typedef __bf16 bf16;
typedef __attribute__((ext_vector_type(16))) __bf16 v16bf;
typedef __attribute__((ext_vector_type(8)))  __bf16 v8bf;
typedef __attribute__((ext_vector_type(8)))  float   v8f;

#define B_  2
#define S_  2048
#define T_  2048
#define E_  1024
#define H_  16
#define D_  64
#define M_  (B_ * S_)   // 4096 rows for every GEMM

__device__ __forceinline__ v16bf combine16(v8bf lo, v8bf hi) {
    v16bf r;
#pragma unroll
    for (int i = 0; i < 8; ++i) { r[i] = lo[i]; r[8 + i] = hi[i]; }
    return r;
}

__device__ __forceinline__ v8f wmma_bf16(v16bf a, v16bf b, v8f c) {
    return __builtin_amdgcn_wmma_f32_16x16x32_bf16(false, a, false, b, (short)0, c,
                                                   false, false);
}

// ---------------------------------------------------------------------------
// Generic GEMM: Out[M,N] = X[M,K] @ W[N,K]^T + bias[N]
// MODE 0: bf16 out, head-major [B,H,S,D], scaled by 1/sqrt(D)   (Q)
// MODE 1: bf16 out, head-major [B,H,T,D]                        (K and V)
// MODE 3: fp32 out, plain row-major [M,N]                       (final proj)
// Workgroup: 256 threads (8 waves), 128x128 output tile.
// Wave (wm = wave>>1, wn = wave&1) owns a 32x64 sub-tile:
//   rows [m0 + 32*wm, +32) (two 16-row WMMA strips),
//   cols [n0 + 64*wn, +64) (four 16-col WMMA strips).
// 8 WMMAs per wave per 32-wide k-step; B fragments reused across M strips.
// ---------------------------------------------------------------------------
template <int MODE, typename TIN>
__global__ __launch_bounds__(256) void gemm_bias_kernel(
    const TIN* __restrict__ X, const float* __restrict__ W,
    const float* __restrict__ bias, void* __restrict__ Out, int K) {
    __shared__ alignas(16) bf16 Al[128 * 32];  // [m][k]
    __shared__ alignas(16) bf16 Bl[128 * 32];  // [n][k]

    const int tid  = threadIdx.x;
    const int lane = tid & 31;
    const int wave = tid >> 5;
    const int hi   = lane >> 4;   // half-wave select
    const int ln   = lane & 15;
    const int wm   = wave >> 1;   // 0..3 : 32-row strip
    const int wn   = wave & 1;    // 0..1 : 64-col strip
    const int m0   = blockIdx.x * 128;
    const int n0   = blockIdx.y * 128;

    v8f acc[2][4] = {};

    for (int k0 = 0; k0 < K; k0 += 32) {
        __syncthreads();
        // stage A tile (convert to bf16): 128x32, 8-element groups
        for (int i = tid; i < 128 * 4; i += 256) {
            int r = i >> 2, c8 = (i & 3) * 8;
            const TIN* src = X + (size_t)(m0 + r) * K + k0 + c8;
            bf16* dst = Al + r * 32 + c8;
#pragma unroll
            for (int j = 0; j < 8; ++j) dst[j] = (bf16)(float)src[j];
        }
        // stage B tile: rows of W are already k-contiguous (x @ W^T)
        for (int i = tid; i < 128 * 4; i += 256) {
            int r = i >> 2, c8 = (i & 3) * 8;
            const float* src = W + (size_t)(n0 + r) * K + k0 + c8;
            bf16* dst = Bl + r * 32 + c8;
#pragma unroll
            for (int j = 0; j < 8; ++j) dst[j] = (bf16)src[j];
        }
        __syncthreads();

        // A fragments: 16x32, lane ln = row, halves at k offsets {hi*8, 16+hi*8}
        v16bf a[2];
#pragma unroll
        for (int mt = 0; mt < 2; ++mt) {
            const bf16* ap = Al + (wm * 32 + mt * 16 + ln) * 32;
            a[mt] = combine16(*(const v8bf*)(ap + hi * 8),
                              *(const v8bf*)(ap + 16 + hi * 8));
        }
#pragma unroll
        for (int nt = 0; nt < 4; ++nt) {
            // B fragment: 32x16, lane ln = col n, halves at k offsets {hi*8, 16+hi*8}
            const bf16* bp = Bl + (wn * 64 + nt * 16 + ln) * 32;
            v16bf b = combine16(*(const v8bf*)(bp + hi * 8), *(const v8bf*)(bp + 16 + hi * 8));
#pragma unroll
            for (int mt = 0; mt < 2; ++mt)
                acc[mt][nt] = wmma_bf16(a[mt], b, acc[mt][nt]);
        }
    }

    // epilogue: bias + layout-specific store
#pragma unroll
    for (int nt = 0; nt < 4; ++nt) {
        int n = n0 + wn * 64 + nt * 16 + ln;
        float bv = bias[n];
#pragma unroll
        for (int mt = 0; mt < 2; ++mt) {
#pragma unroll
            for (int r = 0; r < 8; ++r) {
                int m = m0 + wm * 32 + mt * 16 + r + hi * 8;
                float val = acc[mt][nt][r] + bv;
                if (MODE == 3) {
                    ((float*)Out)[(size_t)m * E_ + n] = val;
                } else {
                    int b = m >> 11, srow = m & 2047;  // m = b*2048 + s
                    int h = n >> 6, d = n & 63;
                    if (MODE == 0) val *= 0.125f;      // fold 1/sqrt(64) into Q
                    ((bf16*)Out)[((size_t)(b * H_ + h) * S_ + srow) * D_ + d] =
                        (bf16)val;
                }
            }
        }
    }
}

// ---------------------------------------------------------------------------
// Flash attention: workgroup = (b,h, 64-row S block), 4 waves x 16 rows.
// Streams 64-wide T chunks; scores + online softmax + P@V all in registers,
// with P transposed through per-wave LDS and V staged transposed in LDS.
// ---------------------------------------------------------------------------
__global__ __launch_bounds__(128) void flash_attn_kernel(
    const bf16* __restrict__ Qb, const bf16* __restrict__ Kb,
    const bf16* __restrict__ Vb, bf16* __restrict__ Cb) {
    __shared__ alignas(16) bf16 Vl[64 * 64];      // [d][t]  (transposed V chunk)
    __shared__ alignas(16) bf16 Pl[4][16 * 64];   // per-wave P staging [m][t]

    const int tid  = threadIdx.x;
    const int lane = tid & 31;
    const int wave = tid >> 5;
    const int hi   = lane >> 4;
    const int ln   = lane & 15;
    const int bh   = blockIdx.x;                  // 0..B*H-1
    const int s0   = blockIdx.y * 64 + wave * 16; // this wave's query rows

    const size_t base = (size_t)bh * T_ * D_;
    const bf16* Qp = Qb + base;
    const bf16* Kp = Kb + base;
    const bf16* Vp = Vb + base;

    // resident Q fragments (16 rows x 64 d = two 16x32 A fragments)
    v16bf qf[2];
#pragma unroll
    for (int kc = 0; kc < 2; ++kc) {
        const bf16* qp = Qp + (size_t)(s0 + ln) * D_ + kc * 32;
        qf[kc] = combine16(*(const v8bf*)(qp + hi * 8),
                           *(const v8bf*)(qp + 16 + hi * 8));
    }

    v8f oacc[4] = {};
    float mrow[8], lrow[8];
#pragma unroll
    for (int r = 0; r < 8; ++r) { mrow[r] = -1e30f; lrow[r] = 0.f; }

    for (int t0 = 0; t0 < T_; t0 += 64) {
        // stage V chunk transposed: Vl[d][t] = V[t0+t][d]
        __syncthreads();
        for (int i = tid; i < 64 * 16; i += 128) {
            int t = i >> 4, d4 = (i & 15) * 4;
            const bf16* vp = Vp + (size_t)(t0 + t) * D_ + d4;
#pragma unroll
            for (int j = 0; j < 4; ++j) Vl[(d4 + j) * 64 + t] = vp[j];
        }
        __syncthreads();

        // scores 16x64: contraction over d; B fragment reads K rows directly
        v8f sc[4];
#pragma unroll
        for (int nt = 0; nt < 4; ++nt) {
            v8f z = {};
#pragma unroll
            for (int kc = 0; kc < 2; ++kc) {
                const bf16* kp =
                    Kp + (size_t)(t0 + nt * 16 + ln) * D_ + kc * 32;
                v16bf kf = combine16(*(const v8bf*)(kp + hi * 8), *(const v8bf*)(kp + 16 + hi * 8));
                z = wmma_bf16(qf[kc], kf, z);
            }
            sc[nt] = z;
        }

        // online softmax: per-row reductions across the 16-lane half
#pragma unroll
        for (int r = 0; r < 8; ++r) {
            float lm = sc[0][r];
#pragma unroll
            for (int nt = 1; nt < 4; ++nt) lm = fmaxf(lm, sc[nt][r]);
#pragma unroll
            for (int off = 8; off; off >>= 1)
                lm = fmaxf(lm, __shfl_xor(lm, off, 16));
            float mnew  = fmaxf(mrow[r], lm);
            float alpha = __expf(mrow[r] - mnew);
            mrow[r] = mnew;
            float ps = 0.f;
#pragma unroll
            for (int nt = 0; nt < 4; ++nt) {
                float e = __expf(sc[nt][r] - mnew);
                sc[nt][r] = e;
                ps += e;
            }
#pragma unroll
            for (int off = 8; off; off >>= 1) ps += __shfl_xor(ps, off, 16);
            lrow[r] = lrow[r] * alpha + ps;
#pragma unroll
            for (int nt = 0; nt < 4; ++nt) oacc[nt][r] *= alpha;
        }

        // transpose P (D-layout -> A-layout) through this wave's LDS slab
        bf16* P = Pl[wave];
#pragma unroll
        for (int nt = 0; nt < 4; ++nt)
#pragma unroll
            for (int r = 0; r < 8; ++r)
                P[(r + hi * 8) * 64 + nt * 16 + ln] = (bf16)sc[nt][r];
        asm volatile("" ::: "memory");  // keep DS stores before DS loads

        // out += P @ V  (contraction over t, two 32-wide k chunks)
#pragma unroll
        for (int kc = 0; kc < 2; ++kc) {
            const bf16* pp = P + (size_t)ln * 64 + kc * 32;
            v16bf af = combine16(*(const v8bf*)(pp + hi * 8),
                                 *(const v8bf*)(pp + 16 + hi * 8));
#pragma unroll
            for (int nt = 0; nt < 4; ++nt) {
                const bf16* vp2 = Vl + (nt * 16 + ln) * 64 + kc * 32;
                v16bf bfr = combine16(*(const v8bf*)(vp2 + hi * 8), *(const v8bf*)(vp2 + 16 + hi * 8));
                oacc[nt] = wmma_bf16(af, bfr, oacc[nt]);
            }
        }
    }

    // normalize and store context in [B,S,E] layout (bf16)
    const int b = bh >> 4, h = bh & 15;
#pragma unroll
    for (int nt = 0; nt < 4; ++nt)
#pragma unroll
        for (int r = 0; r < 8; ++r) {
            float v = oacc[nt][r] / lrow[r];
            int m = s0 + r + hi * 8;
            int d = nt * 16 + ln;
            Cb[(size_t)(b * S_ + m) * E_ + h * D_ + d] = (bf16)v;
        }
}

// ---------------------------------------------------------------------------
extern "C" void kernel_launch(void* const* d_in, const int* in_sizes, int n_in,
                              void* d_out, int out_size, void* d_ws,
                              size_t ws_size, hipStream_t stream) {
    (void)in_sizes; (void)n_in; (void)out_size; (void)ws_size;
    const float* query = (const float*)d_in[0];
    const float* value = (const float*)d_in[1];
    const float* key   = (const float*)d_in[2];
    const float* Wq = (const float*)d_in[3];
    const float* bq = (const float*)d_in[4];
    const float* Wk = (const float*)d_in[5];
    const float* bk = (const float*)d_in[6];
    const float* Wv = (const float*)d_in[7];
    const float* bv = (const float*)d_in[8];
    const float* Wo = (const float*)d_in[9];
    const float* bo = (const float*)d_in[10];

    char* ws = (char*)d_ws;
    const size_t seg = (size_t)M_ * E_ * sizeof(bf16);  // 8 MB each
    bf16* Qb = (bf16*)(ws + 0 * seg);  // [B,H,S,D], pre-scaled by 1/8
    bf16* Kb = (bf16*)(ws + 1 * seg);  // [B,H,T,D]
    bf16* Vb = (bf16*)(ws + 2 * seg);  // [B,H,T,D]
    bf16* Cb = (bf16*)(ws + 3 * seg);  // [B,S,E]

    dim3 blk(256);
    dim3 gproj(M_ / 128, E_ / 128);  // 32 x 8
    gemm_bias_kernel<0, float><<<gproj, blk, 0, stream>>>(query, Wq, bq, Qb, E_);
    gemm_bias_kernel<1, float><<<gproj, blk, 0, stream>>>(key,   Wk, bk, Kb, E_);
    gemm_bias_kernel<1, float><<<gproj, blk, 0, stream>>>(value, Wv, bv, Vb, E_);

    dim3 gattn(B_ * H_, S_ / 64);  // 32 x 32
    flash_attn_kernel<<<gattn, dim3(128), 0, stream>>>(Qb, Kb, Vb, Cb);

    gemm_bias_kernel<3, bf16><<<gproj, blk, 0, stream>>>(Cb, Wo, bo,
                                                         (float*)d_out, E_);
}
